// GIN_encoder_43593918054555
// MI455X (gfx1250) — matrix operand order repaired and checked by reading, not compiled, not run on hardware
//
#include <hip/hip_runtime.h>

typedef __attribute__((ext_vector_type(16))) __bf16 v16bf;
typedef __attribute__((ext_vector_type(8)))  float  v8f;

#define F    256
#define TILE 16

// ---------------- h = x ; zero stats ----------------
__global__ __launch_bounds__(256) void k_init(const float4* __restrict__ x,
                                              float4* __restrict__ h,
                                              float* __restrict__ stats, int total4) {
  int i = blockIdx.x * 256 + threadIdx.x;
  if (i < total4) h[i] = x[i];
  if (i < 1024)   stats[i] = 0.0f;
}

// ---------------- h[dst] += x[src], one wave32 per edge ----------------
__global__ __launch_bounds__(256) void k_scatter(const float* __restrict__ x,
                                                 const long long* __restrict__ ei,
                                                 float* __restrict__ h, int E) {
  int e = blockIdx.x * 8 + (threadIdx.x >> 5);
  if (e >= E) return;
  int lane = threadIdx.x & 31;
  long long s = ei[e];
  long long d = ei[(long long)E + e];
  const float4* xs = (const float4*)(x + s * F) + lane * 2;
  float* hd = h + d * F + lane * 8;
  float4 v0 = xs[0];
  float4 v1 = xs[1];
  unsafeAtomicAdd(hd + 0, v0.x);
  unsafeAtomicAdd(hd + 1, v0.y);
  unsafeAtomicAdd(hd + 2, v0.z);
  unsafeAtomicAdd(hd + 3, v0.w);
  unsafeAtomicAdd(hd + 4, v1.x);
  unsafeAtomicAdd(hd + 5, v1.y);
  unsafeAtomicAdd(hd + 6, v1.z);
  unsafeAtomicAdd(hd + 7, v1.w);
}

__device__ __forceinline__ void cvt8(v16bf& dst, int base, float4 a, float4 b) {
  dst[base + 0] = (__bf16)a.x; dst[base + 1] = (__bf16)a.y;
  dst[base + 2] = (__bf16)a.z; dst[base + 3] = (__bf16)a.w;
  dst[base + 4] = (__bf16)b.x; dst[base + 5] = (__bf16)b.y;
  dst[base + 6] = (__bf16)b.z; dst[base + 7] = (__bf16)b.w;
}

// ---------------- y = h @ W^T + b via v_wmma_f32_16x16x32_bf16 ----------------
// One wave per 16x16 output tile. 50000 rows = 3125 M-tiles exactly, 16 N-tiles.
// A (16x32, bf16): lane m = lane&15; lane<16 holds K 0..7 & 16..23, lane>=16 holds
//                  K 8..15 & 24..31 (ISA 7.12.2 16-bit A table).
// B (32x16, bf16): lane col = lane&15; lane<16 holds K 0..7 & 16..23, lane>=16 holds
//                  K 8..15 & 24..31 (same K order as A).
__global__ __launch_bounds__(256) void k_gemm(const float* __restrict__ h,
                                              const float* __restrict__ W,
                                              const float* __restrict__ bias,
                                              float* __restrict__ y) {
  int lane   = threadIdx.x & 31;
  int tile   = blockIdx.x * 8 + (threadIdx.x >> 5);
  int tm     = tile >> 4;          // M tile (0..3124)
  int tn     = tile & 15;          // N tile (0..15)
  int m      = lane & 15;          // A: M index, B: N (column) index
  int khalfA = (lane >> 4) << 3;   // A: +8 K-offset for upper half-wave
  int khalfB = (lane >> 4) << 3;   // B: +8 K-offset for upper half-wave

  const float* arow = h + (size_t)(tm * TILE + m) * F;  // row of h
  const float* brow = W + (size_t)(tn * TILE + m) * F;  // row of W == column of B
  v8f acc = {};
#pragma unroll
  for (int kk = 0; kk < F; kk += 32) {
    int ka = kk + khalfA;
    int kb = kk + khalfB;
    v16bf av, bv;
    cvt8(av, 0, *(const float4*)(arow + ka),      *(const float4*)(arow + ka + 4));
    cvt8(av, 8, *(const float4*)(arow + ka + 16), *(const float4*)(arow + ka + 20));
    cvt8(bv, 0, *(const float4*)(brow + kb),      *(const float4*)(brow + kb + 4));
    cvt8(bv, 8, *(const float4*)(brow + kb + 16), *(const float4*)(brow + kb + 20));
    acc = __builtin_amdgcn_wmma_f32_16x16x32_bf16(false, av, false, bv,
                                                  (short)0, acc, false, false);
  }
  float bb   = bias[tn * TILE + m];
  int   row0 = tm * TILE + khalfA;  // D: lanes<16 -> M=VGPR i, lanes>=16 -> M=8+i
  int   col  = tn * TILE + m;
#pragma unroll
  for (int i = 0; i < 8; ++i)
    y[(size_t)(row0 + i) * F + col] = acc[i] + bb;
}

// ---------------- per-column sum / sumsq ----------------
__global__ __launch_bounds__(256) void k_stats(const float* __restrict__ y,
                                               float* __restrict__ stats, int N) {
  int c = threadIdx.x;            // one column per thread, coalesced row reads
  float s = 0.f, s2 = 0.f;
  for (int r = blockIdx.x; r < N; r += gridDim.x) {
    float v = y[(size_t)r * F + c];
    s += v; s2 += v * v;
  }
  unsafeAtomicAdd(&stats[c], s);
  unsafeAtomicAdd(&stats[F + c], s2);
}

// ---------------- fold BN stats into scale/shift ----------------
__global__ void k_finalize(float* stats, const float* __restrict__ bnw,
                           const float* __restrict__ bnb, int N) {
  int c = threadIdx.x;
  float inv_n = 1.0f / (float)N;
  float mean = stats[c] * inv_n;
  float var  = stats[F + c] * inv_n - mean * mean;
  float sc   = bnw[c] * rsqrtf(var + 1e-5f);
  stats[2 * F + c] = sc;
  stats[3 * F + c] = bnb[c] - mean * sc;
}

// ---------------- y = y*scale + shift (in place on d_out) ----------------
__global__ __launch_bounds__(256) void k_apply(float4* __restrict__ y,
                                               const float* __restrict__ stats, int total4) {
  int i = blockIdx.x * 256 + threadIdx.x;
  if (i >= total4) return;
  int c = (i & (F / 4 - 1)) * 4;
  const float* sc = stats + 2 * F;
  const float* sh = stats + 3 * F;
  float4 v = y[i];
  v.x = v.x * sc[c + 0] + sh[c + 0];
  v.y = v.y * sc[c + 1] + sh[c + 1];
  v.z = v.z * sc[c + 2] + sh[c + 2];
  v.w = v.w * sc[c + 3] + sh[c + 3];
  y[i] = v;
}

extern "C" void kernel_launch(void* const* d_in, const int* in_sizes, int n_in,
                              void* d_out, int out_size, void* d_ws, size_t ws_size,
                              hipStream_t stream) {
  const float*     x   = (const float*)d_in[0];
  const long long* ei  = (const long long*)d_in[1];
  const float*     W   = (const float*)d_in[2];
  const float*     b   = (const float*)d_in[3];
  const float*     bnw = (const float*)d_in[4];
  const float*     bnb = (const float*)d_in[5];

  int N = in_sizes[0] / F;   // 50000
  int E = in_sizes[1] / 2;   // 1600000

  float* y     = (float*)d_out;
  float* h     = (float*)d_ws;
  float* stats = h + (size_t)N * F;

  int total4 = (N * F) / 4;

  k_init<<<(total4 + 255) / 256, 256, 0, stream>>>((const float4*)x, (float4*)h, stats, total4);
  k_scatter<<<(E + 7) / 8, 256, 0, stream>>>(x, ei, h, E);

  int tiles = (N / TILE) * (F / TILE);  // 50000, divisible by 8
  k_gemm<<<tiles / 8, 256, 0, stream>>>(h, W, b, y);

  k_stats<<<512, 256, 0, stream>>>(y, stats, N);
  k_finalize<<<1, 256, 0, stream>>>(stats, bnw, bnb, N);
  k_apply<<<(total4 + 255) / 256, 256, 0, stream>>>((float4*)y, stats, total4);
}
